// EnhancedTAGAttention_76330158784743
// MI455X (gfx1250) — compile-verified, matrix operand order checked by reading, not run on hardware
//
#include <hip/hip_runtime.h>
#include <hip/hip_bf16.h>

typedef __attribute__((ext_vector_type(16))) _Float16 v16h;
typedef __attribute__((ext_vector_type(8)))  _Float16 v8h;
typedef __attribute__((ext_vector_type(4)))  _Float16 v4h;
typedef __attribute__((ext_vector_type(8)))  float    v8f;
typedef __attribute__((ext_vector_type(4)))  float    v4f;

#define D_MODEL 1024
#define N_HEADS 16
#define HEAD_DIM 64
#define BATCH 2
#define SEQ 2048
#define ROWS (BATCH * SEQ)   // 4096

// ---------------------------------------------------------------------------
// Kernel 1: fused LayerNorm (f32 in) -> f16 out, for query/key/value rows.
// One block per row (1024 elems, 256 threads x 4).
// ---------------------------------------------------------------------------
__global__ __launch_bounds__(256)
void ln_cast_kernel(const float* __restrict__ q, const float* __restrict__ k,
                    const float* __restrict__ v,
                    const float* __restrict__ gamma, const float* __restrict__ beta,
                    _Float16* __restrict__ oq, _Float16* __restrict__ ok,
                    _Float16* __restrict__ ov)
{
    const float* src; _Float16* dst;
    if (blockIdx.y == 0)      { src = q; dst = oq; }
    else if (blockIdx.y == 1) { src = k; dst = ok; }
    else                      { src = v; dst = ov; }

    const size_t row = blockIdx.x;
    const float* x = src + row * D_MODEL;
    float vals[4];
    float s = 0.f, ss = 0.f;
#pragma unroll
    for (int i = 0; i < 4; ++i) {
        float t = x[threadIdx.x + i * 256];
        vals[i] = t; s += t; ss += t * t;
    }
#pragma unroll
    for (int o = 16; o; o >>= 1) { s += __shfl_xor(s, o, 32); ss += __shfl_xor(ss, o, 32); }

    __shared__ float wsum[8], wsq[8];
    const int wave = threadIdx.x >> 5, lane = threadIdx.x & 31;
    if (lane == 0) { wsum[wave] = s; wsq[wave] = ss; }
    __syncthreads();
    s = 0.f; ss = 0.f;
#pragma unroll
    for (int i = 0; i < 8; ++i) { s += wsum[i]; ss += wsq[i]; }

    const float mean = s * (1.0f / D_MODEL);
    const float var  = ss * (1.0f / D_MODEL) - mean * mean;
    const float rstd = rsqrtf(var + 1e-5f);

    _Float16* y = dst + row * D_MODEL;
#pragma unroll
    for (int i = 0; i < 4; ++i) {
        int j = threadIdx.x + i * 256;
        y[j] = (_Float16)((vals[i] - mean) * rstd * gamma[j] + beta[j]);
    }
}

// ---------------------------------------------------------------------------
// Kernel 2: f32 -> f16 cast (weights)
// ---------------------------------------------------------------------------
__global__ __launch_bounds__(256)
void cast_f16_kernel(const float* __restrict__ src, _Float16* __restrict__ dst, int n)
{
    int i = blockIdx.x * 256 + threadIdx.x;
    if (i < n) dst[i] = (_Float16)src[i];
}

// ---------------------------------------------------------------------------
// Kernel 3: C = (A @ W^T + bias) * scale,  A:[M,K] f16, W:[N,K] f16 (row major).
// Both A and B WMMA fragments are contiguous 16-half vectors per lane.
// Block = 8 waves; wave w owns a 16x64 tile (M = blockIdx.x*128 + w*16).
// OUT_F32: 1 -> f32 output (final projection), 0 -> f16 output.
// ---------------------------------------------------------------------------
template <int OUT_F32>
__global__ __launch_bounds__(256)
void gemm_xwT_kernel(const _Float16* __restrict__ A, const _Float16* __restrict__ W,
                     const float* __restrict__ bias, void* __restrict__ Cout,
                     int M, int N, int K, float scale)
{
    const int lane  = threadIdx.x & 31;
    const int wave  = threadIdx.x >> 5;
    const int lmod  = lane & 15;
    const int lhalf = lane >> 4;
    const int m0 = blockIdx.x * 128 + wave * 16;
    const int n0 = blockIdx.y * 64;

    v8f acc[4] = {};
    const _Float16* arow = A + (size_t)(m0 + lmod) * K + lhalf * 16;

    for (int k0 = 0; k0 < K; k0 += 32) {
        v16h afrag = *(const v16h*)(arow + k0);
#pragma unroll
        for (int t = 0; t < 4; ++t) {
            const _Float16* brow = W + (size_t)(n0 + t * 16 + lmod) * K + k0 + lhalf * 16;
            v16h bfrag = *(const v16h*)brow;
            acc[t] = __builtin_amdgcn_wmma_f32_16x16x32_f16(
                false, afrag, false, bfrag, (short)0, acc[t], false, false);
        }
    }

#pragma unroll
    for (int t = 0; t < 4; ++t) {
        const int n = n0 + t * 16 + lmod;
        const float bval = bias[n];
#pragma unroll
        for (int vv = 0; vv < 8; ++vv) {
            const int m = m0 + vv + lhalf * 8;
            float r = (acc[t][vv] + bval) * scale;
            if (OUT_F32) ((float*)Cout)[(size_t)m * N + n] = r;
            else         ((_Float16*)Cout)[(size_t)m * N + n] = (_Float16)r;
        }
    }
}

// ---------------------------------------------------------------------------
// Kernel 4: transpose V (f16) from [b, k, h*64+d] to Vt [b, h, d, k] so PV
// B-fragments become contiguous 32B vector loads. LDS-tiled, coalesced both
// sides. Grid: (SEQ/64, H, B), block 256.
// ---------------------------------------------------------------------------
__global__ __launch_bounds__(256)
void transpose_v_kernel(const _Float16* __restrict__ Vh, _Float16* __restrict__ Vt)
{
    __shared__ _Float16 tile[64][72];   // [k][d], padded to dodge bank conflicts
    const int k0 = blockIdx.x * 64;
    const int h  = blockIdx.y;
    const int b  = blockIdx.z;

#pragma unroll
    for (int pass = 0; pass < 2; ++pass) {
        const int i  = (threadIdx.x >> 3) + pass * 32;   // k row in tile
        const int dc = (threadIdx.x & 7) * 8;            // d chunk
        const _Float16* src =
            Vh + (size_t)(b * SEQ + k0 + i) * D_MODEL + h * HEAD_DIM + dc;
        v8h val = *(const v8h*)src;
#pragma unroll
        for (int j = 0; j < 8; ++j) tile[i][dc + j] = val[j];
    }
    __syncthreads();

    const int d  = threadIdx.x >> 2;          // 0..63
    const int kc = (threadIdx.x & 3) * 16;    // 0,16,32,48
    _Float16* dst =
        Vt + ((size_t)(b * N_HEADS + h) * HEAD_DIM + d) * SEQ + k0 + kc;
    v16h outv;
#pragma unroll
    for (int j = 0; j < 16; ++j) outv[j] = tile[kc + j][d];
    *(v16h*)dst = outv;
}

// ---------------------------------------------------------------------------
// Kernel 5: attention core. One block per (b, h, 16-query tile).
// LDS: f32 scores [16][2048] (128KB) + f16 probs [16][2048] (64KB) + PV scratch.
// Phase 1: scores via WMMA + relation bias (NT loads) + branchless mask -> LDS.
// Phase 2: row softmax (wave32 shuffles, b128 LDS passes), NT f32 attn_weights,
//          f16 probs -> LDS.
// Phase 3: PV via WMMA (probs + transposed V, both contiguous frags).
// ---------------------------------------------------------------------------
__global__ __launch_bounds__(256)
void attn_kernel(const _Float16* __restrict__ Qh, const _Float16* __restrict__ Kh,
                 const _Float16* __restrict__ Vt,
                 const int* __restrict__ mask, const float* __restrict__ tagrel,
                 const float* __restrict__ rel_type_w, const float* __restrict__ rel_scale,
                 float* __restrict__ attn_w, _Float16* __restrict__ ctx)
{
    extern __shared__ char smem[];
    float*    sc = (float*)smem;                                  // [16][2048] f32
    _Float16* ph = (_Float16*)(smem + 16 * SEQ * 4);              // [16][2048] f16
    float*    pv = (float*)(smem + 16 * SEQ * 4 + 16 * SEQ * 2);  // [8][256] f32

    const int q0 = blockIdx.x * 16;
    const int h  = blockIdx.y;
    const int b  = blockIdx.z;
    const int lane  = threadIdx.x & 31;
    const int wave  = threadIdx.x >> 5;
    const int lmod  = lane & 15;
    const int lhalf = lane >> 4;

    // relation softmax weights + sigmoid(rel_scale)*0.1 (cheap, per-thread)
    const float w0 = rel_type_w[0], w1 = rel_type_w[1], w2 = rel_type_w[2], w3 = rel_type_w[3];
    const float wmx = fmaxf(fmaxf(w0, w1), fmaxf(w2, w3));
    const float e0 = __expf(w0 - wmx), e1 = __expf(w1 - wmx),
                e2 = __expf(w2 - wmx), e3 = __expf(w3 - wmx);
    const float einv = 1.0f / (e0 + e1 + e2 + e3);
    const float rw0 = e0 * einv, rw1 = e1 * einv, rw2 = e2 * einv, rw3 = e3 * einv;
    const float coef = 0.1f / (1.0f + __expf(-rel_scale[0]));

    // Q fragments for this 16-row tile (1/sqrt(64) already folded into Qh)
    const _Float16* qrow =
        Qh + (size_t)(b * SEQ + q0 + lmod) * D_MODEL + h * HEAD_DIM + lhalf * 16;
    v16h aq0 = *(const v16h*)qrow;
    v16h aq1 = *(const v16h*)(qrow + 32);

    // ---- Phase 1: scores ----
    for (int kt = wave; kt < SEQ / 16; kt += 8) {
        const int k0 = kt * 16;
        const _Float16* krow =
            Kh + (size_t)(b * SEQ + k0 + lmod) * D_MODEL + h * HEAD_DIM + lhalf * 16;
        // prefetch next K tile for this wave (global_prefetch_b8)
        if (kt + 8 < SEQ / 16)
            __builtin_prefetch(krow + (size_t)128 * D_MODEL, 0, 1);

        v8f accv = {};
        v16h bk0 = *(const v16h*)krow;
        v16h bk1 = *(const v16h*)(krow + 32);
        accv = __builtin_amdgcn_wmma_f32_16x16x32_f16(false, aq0, false, bk0,
                                                      (short)0, accv, false, false);
        accv = __builtin_amdgcn_wmma_f32_16x16x32_f16(false, aq1, false, bk1,
                                                      (short)0, accv, false, false);

        const int n = k0 + lmod;                        // key index for this lane
        const int mk = mask[b * SEQ + n];
        const v4f* trbase = (const v4f*)tagrel + (size_t)(b * SEQ + q0) * SEQ + n;
#pragma unroll
        for (int vv = 0; vv < 8; ++vv) {
            const int m = vv + lhalf * 8;               // query row in tile
            v4f t4 = __builtin_nontemporal_load(trbase + (size_t)m * SEQ);
            float wsumr = t4[0] * rw0 + t4[1] * rw1 + t4[2] * rw2 + t4[3] * rw3;
            float sval = accv[vv] + wsumr * coef;
            sc[m * SEQ + n] = (mk == 0) ? -1e9f : sval;  // branchless select
        }
    }
    __syncthreads();

    // ---- Phase 2: softmax (wave w handles rows 2w, 2w+1), 16B vectorized ----
    for (int r = wave * 2; r < wave * 2 + 2; ++r) {
        float* row = sc + r * SEQ;
        float mx = -3.4e38f;
        for (int j = lane * 4; j < SEQ; j += 128) {
            v4f t = *(const v4f*)(row + j);
            mx = fmaxf(mx, fmaxf(fmaxf(t[0], t[1]), fmaxf(t[2], t[3])));
        }
#pragma unroll
        for (int o = 16; o; o >>= 1) mx = fmaxf(mx, __shfl_xor(mx, o, 32));

        float sum = 0.f;
        for (int j = lane * 4; j < SEQ; j += 128) {
            v4f t = *(const v4f*)(row + j);
            v4f e;
            e[0] = __expf(t[0] - mx); e[1] = __expf(t[1] - mx);
            e[2] = __expf(t[2] - mx); e[3] = __expf(t[3] - mx);
            *(v4f*)(row + j) = e;
            sum += e[0] + e[1] + e[2] + e[3];
        }
#pragma unroll
        for (int o = 16; o; o >>= 1) sum += __shfl_xor(sum, o, 32);
        const float inv = 1.0f / sum;

        float* gw = attn_w + ((((size_t)b * N_HEADS + h) * SEQ) + (q0 + r)) * SEQ;
        _Float16* prow = ph + r * SEQ;
        for (int j = lane * 4; j < SEQ; j += 128) {
            v4f t = *(const v4f*)(row + j);
            v4f p = t * inv;
            __builtin_nontemporal_store(p, (v4f*)(gw + j));   // streaming f32 out
            v4h p16;
            p16[0] = (_Float16)p[0]; p16[1] = (_Float16)p[1];
            p16[2] = (_Float16)p[2]; p16[3] = (_Float16)p[3];
            *(v4h*)(prow + j) = p16;                          // f16 probs for PV
        }
    }
    __syncthreads();

    // ---- Phase 3: PV.  wave -> (d-tile = wave&3, k-half = wave>>2) ----
    {
        const int dt = wave & 3;
        const int kh = wave >> 2;
        v8f accp = {};
        const _Float16* vtrow =
            Vt + ((size_t)(b * N_HEADS + h) * HEAD_DIM + dt * 16 + lmod) * SEQ;
        for (int k0 = kh * (SEQ / 2); k0 < kh * (SEQ / 2) + SEQ / 2; k0 += 32) {
            v16h ap = *(const v16h*)(ph + lmod * SEQ + k0 + lhalf * 16);
            v16h bvf = *(const v16h*)(vtrow + k0 + lhalf * 16);
            accp = __builtin_amdgcn_wmma_f32_16x16x32_f16(false, ap, false, bvf,
                                                          (short)0, accp, false, false);
        }
#pragma unroll
        for (int vv = 0; vv < 8; ++vv) pv[wave * 256 + vv * 32 + lane] = accp[vv];
    }
    __syncthreads();

    // combine the two k-halves, write f16 context [b, q, h*64 + d]
#pragma unroll
    for (int i = 0; i < 4; ++i) {
        const int idx = threadIdx.x + i * 256;
        const int dtile = idx >> 8;
        const int r = idx & 255;
        const int vv = r >> 5;
        const int ln = r & 31;
        const float s = pv[dtile * 256 + r] + pv[(dtile + 4) * 256 + r];
        const int m = vv + ((ln >> 4) << 3);
        const int n = ln & 15;
        ctx[(size_t)(b * SEQ + q0 + m) * D_MODEL + h * HEAD_DIM + dtile * 16 + n] = (_Float16)s;
    }
}

// ---------------------------------------------------------------------------
// Host launcher
// ---------------------------------------------------------------------------
extern "C" void kernel_launch(void* const* d_in, const int* in_sizes, int n_in,
                              void* d_out, int out_size, void* d_ws, size_t ws_size,
                              hipStream_t stream)
{
    const float* query  = (const float*)d_in[0];
    const float* key    = (const float*)d_in[1];
    const float* value  = (const float*)d_in[2];
    const int*   mask   = (const int*)  d_in[3];
    const float* tagrel = (const float*)d_in[4];
    const float* gamma  = (const float*)d_in[5];
    const float* beta   = (const float*)d_in[6];
    const float* Wq = (const float*)d_in[7];   const float* bq = (const float*)d_in[8];
    const float* Wk = (const float*)d_in[9];   const float* bk = (const float*)d_in[10];
    const float* Wv = (const float*)d_in[11];  const float* bv = (const float*)d_in[12];
    const float* Wo = (const float*)d_in[13];  const float* bo = (const float*)d_in[14];
    const float* relw   = (const float*)d_in[15];
    const float* relsc  = (const float*)d_in[16];

    char* ws = (char*)d_ws;
    const size_t MB = 1u << 20;
    _Float16* XQh = (_Float16*)(ws + 0 * MB);   // LN(q) f16, 8MB
    _Float16* XKh = (_Float16*)(ws + 8 * MB);
    _Float16* XVh = (_Float16*)(ws + 16 * MB);
    _Float16* WQh = (_Float16*)(ws + 24 * MB);  // f16 weights, 2MB each
    _Float16* WKh = (_Float16*)(ws + 26 * MB);
    _Float16* WVh = (_Float16*)(ws + 28 * MB);
    _Float16* WOh = (_Float16*)(ws + 30 * MB);
    _Float16* Qh  = (_Float16*)(ws + 32 * MB);  // projected, f16, 8MB each
    _Float16* Kh  = (_Float16*)(ws + 40 * MB);
    _Float16* Vh  = (_Float16*)(ws + 48 * MB);
    _Float16* Ctx = (_Float16*)(ws + 56 * MB);
    _Float16* Vt  = (_Float16*)(ws + 64 * MB);  // transposed V [b,h,d,k], 8MB

    float* out_attn = (float*)d_out;                               // [B,S,D]
    float* out_w    = (float*)d_out + (size_t)ROWS * D_MODEL;      // [B,H,S,S]

    // 1) LayerNorm + cast for q/k/v
    ln_cast_kernel<<<dim3(ROWS, 3), 256, 0, stream>>>(query, key, value, gamma, beta,
                                                      XQh, XKh, XVh);
    // 2) weight casts
    const int WN = D_MODEL * D_MODEL;
    cast_f16_kernel<<<(WN + 255) / 256, 256, 0, stream>>>(Wq, WQh, WN);
    cast_f16_kernel<<<(WN + 255) / 256, 256, 0, stream>>>(Wk, WKh, WN);
    cast_f16_kernel<<<(WN + 255) / 256, 256, 0, stream>>>(Wv, WVh, WN);
    cast_f16_kernel<<<(WN + 255) / 256, 256, 0, stream>>>(Wo, WOh, WN);

    // 3) projections (scale 1/sqrt(64) folded into Q)
    dim3 ggrid(ROWS / 128, D_MODEL / 64);
    gemm_xwT_kernel<0><<<ggrid, 256, 0, stream>>>(XQh, WQh, bq, (void*)Qh,
                                                  ROWS, D_MODEL, D_MODEL, 0.125f);
    gemm_xwT_kernel<0><<<ggrid, 256, 0, stream>>>(XKh, WKh, bk, (void*)Kh,
                                                  ROWS, D_MODEL, D_MODEL, 1.0f);
    gemm_xwT_kernel<0><<<ggrid, 256, 0, stream>>>(XVh, WVh, bv, (void*)Vh,
                                                  ROWS, D_MODEL, D_MODEL, 1.0f);

    // 4) transpose V for contiguous PV fragments
    transpose_v_kernel<<<dim3(SEQ / 64, N_HEADS, BATCH), 256, 0, stream>>>(Vh, Vt);

    // 5) attention core (dynamic LDS: 128KB scores + 64KB probs + 8KB PV scratch)
    const size_t smem = (size_t)16 * SEQ * 4 + (size_t)16 * SEQ * 2 + 8 * 256 * 4;
    attn_kernel<<<dim3(SEQ / 16, N_HEADS, BATCH), 256, smem, stream>>>(
        Qh, Kh, Vt, mask, tagrel, relw, relsc, out_w, Ctx);

    // 6) output projection -> f32
    gemm_xwT_kernel<1><<<ggrid, 256, 0, stream>>>(Ctx, WOh, bo, (void*)out_attn,
                                                  ROWS, D_MODEL, D_MODEL, 1.0f);
}
